// PointNet2FeatureExtractorWithFP_ShapeNet_6700148982227
// MI455X (gfx1250) — matrix operand order repaired and checked by reading, not compiled, not run on hardware
//
#include <hip/hip_runtime.h>
#include <hip/hip_bf16.h>
#include <cstdint>
#include <cstddef>

typedef __attribute__((ext_vector_type(16))) _Float16 v16h;
typedef __attribute__((ext_vector_type(8)))  _Float16 v8h;
typedef __attribute__((ext_vector_type(8)))  float    v8f;

#define B    16
#define NPTS 8192
#define WPB  4   // waves per block in the WMMA GEMM

// ---------------------------------------------------------------- FPS
// One block (256 threads) per batch; serial over S iterations with a
// block-wide argmax reduction. Emitted index per iteration is the incoming
// farthest point (jax.lax.scan carry semantics).
__global__ void k_fps(const float* __restrict__ xyz, int N, int S,
                      int* __restrict__ fidx, float* __restrict__ dist)
{
    int b = blockIdx.x;
    const float* p = xyz + (size_t)b * N * 3;
    float* d = dist + (size_t)b * N;
    __shared__ float sval[256];
    __shared__ int   sidx[256];
    __shared__ int   sfar;
    int t = threadIdx.x;
    for (int i = t; i < N; i += 256) d[i] = 1e10f;
    if (t == 0) sfar = 0;
    __syncthreads();
    for (int it = 0; it < S; ++it) {
        int far = sfar;
        if (t == 0) fidx[b * S + it] = far;
        float cx = p[far*3+0], cy = p[far*3+1], cz = p[far*3+2];
        float best = -1.0f; int bi = 0;
        for (int i = t; i < N; i += 256) {
            float dx = p[i*3+0]-cx, dy = p[i*3+1]-cy, dz = p[i*3+2]-cz;
            float dd = dx*dx + dy*dy + dz*dz;
            float m = fminf(d[i], dd);
            d[i] = m;
            if (m > best) { best = m; bi = i; }
        }
        sval[t] = best; sidx[t] = bi;
        __syncthreads();
        for (int s = 128; s > 0; s >>= 1) {
            if (t < s) {
                float ov = sval[t+s]; int oi = sidx[t+s];
                if (ov > sval[t] || (ov == sval[t] && oi < sidx[t])) {
                    sval[t] = ov; sidx[t] = oi;
                }
            }
            __syncthreads();
        }
        if (t == 0) sfar = sidx[0];
        __syncthreads();
    }
}

// ---------------------------------------------------------------- gather xyz[fidx]
__global__ void k_gather(const float* __restrict__ xyz, const int* __restrict__ fidx,
                         int N, int S, float* __restrict__ out)
{
    int tid = blockIdx.x * blockDim.x + threadIdx.x;
    if (tid >= B * S) return;
    int b = tid / S, s = tid % S;
    int j = fidx[b * S + s];
    const float* p = xyz + ((size_t)b * N + j) * 3;
    float* o = out + ((size_t)b * S + s) * 3;
    o[0] = p[0]; o[1] = p[1]; o[2] = p[2];
}

// ---------------------------------------------------------------- ball query
__global__ void k_ballquery(const float* __restrict__ xyz, const float* __restrict__ nxyz,
                            int N, int S, int K, float r2, int* __restrict__ idx)
{
    int tid = blockIdx.x * blockDim.x + threadIdx.x;
    if (tid >= B * S) return;
    int b = tid / S, s = tid % S;
    const float* p = xyz + (size_t)b * N * 3;
    const float* q = nxyz + ((size_t)b * S + s) * 3;
    float qx = q[0], qy = q[1], qz = q[2];
    int* o = idx + ((size_t)b * S + s) * K;
    int cnt = 0, first = 0; bool found = false;
    for (int j = 0; j < N && cnt < K; ++j) {
        float dx = p[j*3+0]-qx, dy = p[j*3+1]-qy, dz = p[j*3+2]-qz;
        float dd = dx*dx + dy*dy + dz*dz;
        if (dd <= r2) {
            if (!found) { found = true; first = j; }
            o[cnt++] = j;
        }
    }
    for (; cnt < K; ++cnt) o[cnt] = first;
}

// ---------------------------------------------------------------- grouping -> f16 X (K padded)
// X row (b,s,k) = [xyz[idx]-center | feats[idx] | 0-pad] ; Cpad % 32 == 0
__global__ void k_group(const float* __restrict__ xyz, const float* __restrict__ nxyz,
                        const _Float16* __restrict__ feats, const int* __restrict__ idx,
                        int N, int S, int K, int Cf, int Cpad, _Float16* __restrict__ X)
{
    int tid = blockIdx.x * blockDim.x + threadIdx.x;
    if (tid >= B * S * K) return;
    int k = tid % K;
    int s = (tid / K) % S;
    int b = tid / (K * S);
    int j = idx[((size_t)b * S + s) * K + k];
    const float* p = xyz + ((size_t)b * N + j) * 3;
    const float* q = nxyz + ((size_t)b * S + s) * 3;
    _Float16* o = X + (((size_t)b * S + s) * K + k) * Cpad;
    o[0] = (_Float16)(p[0] - q[0]);
    o[1] = (_Float16)(p[1] - q[1]);
    o[2] = (_Float16)(p[2] - q[2]);
    if (feats) {
        const _Float16* f = feats + ((size_t)b * N + j) * Cf;
        for (int c = 0; c < Cf; ++c) o[3 + c] = f[c];
    }
    for (int c = 3 + Cf; c < Cpad; ++c) o[c] = (_Float16)0.0f;
}

// ---------------------------------------------------------------- weight prep: f32 (O,C) -> f16 (O,Cpad)
__global__ void k_prep_w(const float* __restrict__ W, _Float16* __restrict__ Wh,
                         int O, int C, int Cpad)
{
    int tid = blockIdx.x * blockDim.x + threadIdx.x;
    if (tid >= O * Cpad) return;
    int o = tid / Cpad, c = tid % Cpad;
    Wh[tid] = (_Float16)(c < C ? W[(size_t)o * C + c] : 0.0f);
}

// ---------------------------------------------------------------- WMMA GEMM
// Y[r,o] = relu( X[r,:] . Wh[o,:] + bias[o] ), all f16 in / f16 out, f32 accum.
// Preconditions: R % 64 == 0, C % 32 == 0, O % 64 == 0 (no bounds checks).
// Each wave computes a 16x64 strip: A fragment reused across 4 WMMAs/K-chunk.
union AF { v16h v; v8h h8[2]; };
union CF { v8f  v; float f[8]; };

__global__ void __launch_bounds__(WPB * 32)
k_gemm(const _Float16* __restrict__ X, const _Float16* __restrict__ Wh,
       const float* __restrict__ bias, _Float16* __restrict__ Y,
       int R, int C, int O)
{
    int wave = threadIdx.x >> 5;
    int lane = threadIdx.x & 31;
    int l  = lane & 15;
    int hi = lane >> 4;
    int row0 = (blockIdx.x * WPB + wave) * 16;
    int col0 = blockIdx.y * 64;

    const _Float16* xr = X + ((size_t)(row0 + l)) * C;      // A row for this lane
    const _Float16* w0 = Wh + ((size_t)(col0 + l)) * C;     // B columns (W rows)
    const _Float16* w1 = w0 + (size_t)16 * C;
    const _Float16* w2 = w0 + (size_t)32 * C;
    const _Float16* w3 = w0 + (size_t)48 * C;

    float bias0 = bias[col0 + l];
    float bias1 = bias[col0 + 16 + l];
    float bias2 = bias[col0 + 32 + l];
    float bias3 = bias[col0 + 48 + l];

    v8f acc0 = {}, acc1 = {}, acc2 = {}, acc3 = {};
    for (int k0 = 0; k0 < C; k0 += 32) {
        // A fragment: 16 contiguous f16 at k0 + hi*16 (same k order as B)
        AF a;
        a.h8[0] = *(const v8h*)(xr + k0 + hi * 16);
        a.h8[1] = *(const v8h*)(xr + k0 + hi * 16 + 8);
        // B fragments (contiguous k order, same as A): 16 contiguous f16 at k0 + hi*16
        int kb = k0 + hi * 16;
        AF b0, b1, b2, b3;
        b0.h8[0] = *(const v8h*)(w0 + kb); b0.h8[1] = *(const v8h*)(w0 + kb + 8);
        b1.h8[0] = *(const v8h*)(w1 + kb); b1.h8[1] = *(const v8h*)(w1 + kb + 8);
        b2.h8[0] = *(const v8h*)(w2 + kb); b2.h8[1] = *(const v8h*)(w2 + kb + 8);
        b3.h8[0] = *(const v8h*)(w3 + kb); b3.h8[1] = *(const v8h*)(w3 + kb + 8);
        acc0 = __builtin_amdgcn_wmma_f32_16x16x32_f16(false, a.v, false, b0.v, (short)0, acc0, false, false);
        acc1 = __builtin_amdgcn_wmma_f32_16x16x32_f16(false, a.v, false, b1.v, (short)0, acc1, false, false);
        acc2 = __builtin_amdgcn_wmma_f32_16x16x32_f16(false, a.v, false, b2.v, (short)0, acc2, false, false);
        acc3 = __builtin_amdgcn_wmma_f32_16x16x32_f16(false, a.v, false, b3.v, (short)0, acc3, false, false);
    }

    CF d0, d1, d2, d3;
    d0.v = acc0; d1.v = acc1; d2.v = acc2; d3.v = acc3;
    #pragma unroll
    for (int v = 0; v < 8; ++v) {
        int rr = row0 + v + 8 * hi;                 // C/D layout: lanes 16-31 hold M=8..15
        _Float16* yr = Y + (size_t)rr * O + col0 + l;
        yr[0]  = (_Float16)fmaxf(d0.f[v] + bias0, 0.0f);
        yr[16] = (_Float16)fmaxf(d1.f[v] + bias1, 0.0f);
        yr[32] = (_Float16)fmaxf(d2.f[v] + bias2, 0.0f);
        yr[48] = (_Float16)fmaxf(d3.f[v] + bias3, 0.0f);
    }
}

// ---------------------------------------------------------------- max over nsample (f16 -> f16)
__global__ void k_maxpool(const _Float16* __restrict__ X, _Float16* __restrict__ out,
                          int groups, int K, int C)
{
    int tid = blockIdx.x * blockDim.x + threadIdx.x;
    if (tid >= groups * C) return;
    int g = tid / C, c = tid % C;
    const _Float16* p = X + (size_t)g * K * C + c;
    float m = (float)p[0];
    for (int k = 1; k < K; ++k) m = fmaxf(m, (float)p[(size_t)k * C]);
    out[(size_t)g * C + c] = (_Float16)m;
}

// ---------------------------------------------------------------- 3-NN interp + concat -> f16 X
__global__ void k_interp3(const float* __restrict__ xyz1, const float* __restrict__ xyz2,
                          const _Float16* __restrict__ feats2, const _Float16* __restrict__ feats1,
                          int S1, int S2, int C2, int C1, _Float16* __restrict__ X)
{
    int tid = blockIdx.x * blockDim.x + threadIdx.x;
    if (tid >= B * S1) return;
    int b = tid / S1, n = tid % S1;
    const float* q = xyz1 + ((size_t)b * S1 + n) * 3;
    const float* p = xyz2 + (size_t)b * S2 * 3;
    float qx = q[0], qy = q[1], qz = q[2];
    float d0 = 1e30f, d1 = 1e30f, d2 = 1e30f;
    int   i0 = 0,     i1 = 0,     i2 = 0;
    for (int j = 0; j < S2; ++j) {
        float dx = p[j*3+0]-qx, dy = p[j*3+1]-qy, dz = p[j*3+2]-qz;
        float dd = dx*dx + dy*dy + dz*dz;
        if (dd < d0)      { d2=d1; i2=i1; d1=d0; i1=i0; d0=dd; i0=j; }
        else if (dd < d1) { d2=d1; i2=i1; d1=dd; i1=j; }
        else if (dd < d2) { d2=dd; i2=j; }
    }
    float w0 = 1.0f / (d0 + 1e-8f);
    float w1 = 1.0f / (d1 + 1e-8f);
    float w2 = 1.0f / (d2 + 1e-8f);
    float ws = w0 + w1 + w2;
    w0 /= ws; w1 /= ws; w2 /= ws;
    int C = C2 + C1;
    _Float16* o = X + ((size_t)b * S1 + n) * C;
    const _Float16* f2 = feats2 + (size_t)b * S2 * C2;
    for (int c = 0; c < C2; ++c)
        o[c] = (_Float16)(w0 * (float)f2[(size_t)i0 * C2 + c]
                        + w1 * (float)f2[(size_t)i1 * C2 + c]
                        + w2 * (float)f2[(size_t)i2 * C2 + c]);
    if (feats1) {
        const _Float16* f1 = feats1 + ((size_t)b * S1 + n) * C1;
        for (int c = 0; c < C1; ++c) o[C2 + c] = f1[c];
    }
}

// ---------------------------------------------------------------- (B,N,C) f16 -> (B,C,N) f32
__global__ void k_transpose(const _Float16* __restrict__ X, float* __restrict__ out,
                            int N, int C)
{
    size_t tid = (size_t)blockIdx.x * blockDim.x + threadIdx.x;
    size_t total = (size_t)B * N * C;
    if (tid >= total) return;
    int c = (int)(tid % C);
    int n = (int)((tid / C) % N);
    int b = (int)(tid / ((size_t)N * C));
    out[((size_t)b * C + c) * N + n] = (float)X[tid];
}

// ================================================================ host side
static inline int cdiv(int a, int b) { return (a + b - 1) / b; }

static void gemm(const _Float16* X, const _Float16* Wh, const float* bias, _Float16* Y,
                 int R, int C, int O, hipStream_t s)
{
    dim3 grid(R / (16 * WPB), O / 64);
    k_gemm<<<grid, WPB * 32, 0, s>>>(X, Wh, bias, Y, R, C, O);
}

extern "C" void kernel_launch(void* const* d_in, const int* in_sizes, int n_in,
                              void* d_out, int out_size, void* d_ws, size_t ws_size,
                              hipStream_t stream)
{
    const float* xyz = (const float*)d_in[0];
    const float* P[30];
    for (int i = 0; i < 30 && i + 1 < n_in; ++i) P[i] = (const float*)d_in[i + 1];

    // ---- workspace carve
    char* wp = (char*)d_ws;
    auto carve = [&](size_t nbytes) {
        void* r = (void*)wp;
        wp += (nbytes + 255) & ~(size_t)255;
        return r;
    };
    float* dist = (float*)carve((size_t)B * NPTS * 4);
    int*   fidx = (int*)  carve((size_t)B * 512 * 4);
    int*   idxb = (int*)  carve((size_t)B * 512 * 32 * 4);       // max over levels
    float* l1x  = (float*)carve((size_t)B * 512 * 3 * 4);
    float* l2x  = (float*)carve((size_t)B * 128 * 3 * 4);
    float* l3x  = (float*)carve((size_t)B * 32  * 3 * 4);
    _Float16* l1f = (_Float16*)carve((size_t)B * 512 * 128 * 2);
    _Float16* l2f = (_Float16*)carve((size_t)B * 128 * 256 * 2);
    _Float16* l3f = (_Float16*)carve((size_t)B * 32  * 512 * 2);
    _Float16* l2u = (_Float16*)carve((size_t)B * 128 * 256 * 2);
    _Float16* l1u = (_Float16*)carve((size_t)B * 512 * 128 * 2);

    // padded f16 weights: {O, C, Cpad} per layer (param i -> W=P[2i], b=P[2i+1])
    const int LW[15][3] = {
        { 64,   3,  32}, { 64,  64,  64}, {128,  64,  64},      // sa1
        {128, 131, 160}, {128, 128, 128}, {256, 128, 128},      // sa2
        {256, 259, 288}, {256, 256, 256}, {512, 256, 256},      // sa3
        {512, 768, 768}, {256, 512, 512},                       // fp3
        {256, 384, 384}, {128, 256, 256},                       // fp2
        {128, 128, 128}, {256, 128, 128}                        // fp1
    };
    _Float16* Wh[15];
    for (int i = 0; i < 15; ++i)
        Wh[i] = (_Float16*)carve((size_t)LW[i][0] * LW[i][2] * 2);

    const size_t BIG = (size_t)33554432;                         // 32M f16 ping-pong
    _Float16* bufA = (_Float16*)carve(BIG * 2);
    _Float16* bufB = (_Float16*)carve(BIG * 2);
    (void)ws_size; (void)in_sizes; (void)out_size;

    const int TB = 256;

    // ---- convert/pad all weights to f16
    for (int i = 0; i < 15; ++i) {
        int tot = LW[i][0] * LW[i][2];
        k_prep_w<<<cdiv(tot, TB), TB, 0, stream>>>(P[2*i], Wh[i], LW[i][0], LW[i][1], LW[i][2]);
    }

    // ================= SA1: 8192 -> 512, K=32, r=0.2, MLP (3|32)->64->64->128
    {
        int N = NPTS, S = 512, K = 32, R = B * S * K;
        k_fps<<<B, 256, 0, stream>>>(xyz, N, S, fidx, dist);
        k_gather<<<cdiv(B*S, TB), TB, 0, stream>>>(xyz, fidx, N, S, l1x);
        k_ballquery<<<cdiv(B*S, TB), TB, 0, stream>>>(xyz, l1x, N, S, K, 0.04f, idxb);
        k_group<<<cdiv(B*S*K, TB), TB, 0, stream>>>(xyz, l1x, nullptr, idxb, N, S, K, 0, 32, bufA);
        gemm(bufA, Wh[0], P[1], bufB, R, 32, 64, stream);
        gemm(bufB, Wh[1], P[3], bufA, R, 64, 64, stream);
        gemm(bufA, Wh[2], P[5], bufB, R, 64, 128, stream);
        k_maxpool<<<cdiv(B*S*128, TB), TB, 0, stream>>>(bufB, l1f, B*S, K, 128);
    }
    // ================= SA2: 512 -> 128, K=64, r=0.4, MLP (131|160)->128->128->256
    {
        int N = 512, S = 128, K = 64, R = B * S * K;
        k_fps<<<B, 256, 0, stream>>>(l1x, N, S, fidx, dist);
        k_gather<<<cdiv(B*S, TB), TB, 0, stream>>>(l1x, fidx, N, S, l2x);
        k_ballquery<<<cdiv(B*S, TB), TB, 0, stream>>>(l1x, l2x, N, S, K, 0.16f, idxb);
        k_group<<<cdiv(B*S*K, TB), TB, 0, stream>>>(l1x, l2x, l1f, idxb, N, S, K, 128, 160, bufA);
        gemm(bufA, Wh[3], P[7],  bufB, R, 160, 128, stream);
        gemm(bufB, Wh[4], P[9],  bufA, R, 128, 128, stream);
        gemm(bufA, Wh[5], P[11], bufB, R, 128, 256, stream);
        k_maxpool<<<cdiv(B*S*256, TB), TB, 0, stream>>>(bufB, l2f, B*S, K, 256);
    }
    // ================= SA3: 128 -> 32, K=128, r=0.6, MLP (259|288)->256->256->512
    {
        int N = 128, S = 32, K = 128, R = B * S * K;
        k_fps<<<B, 256, 0, stream>>>(l2x, N, S, fidx, dist);
        k_gather<<<cdiv(B*S, TB), TB, 0, stream>>>(l2x, fidx, N, S, l3x);
        k_ballquery<<<cdiv(B*S, TB), TB, 0, stream>>>(l2x, l3x, N, S, K, 0.36f, idxb);
        k_group<<<cdiv(B*S*K, TB), TB, 0, stream>>>(l2x, l3x, l2f, idxb, N, S, K, 256, 288, bufA);
        gemm(bufA, Wh[6], P[13], bufB, R, 288, 256, stream);
        gemm(bufB, Wh[7], P[15], bufA, R, 256, 256, stream);
        gemm(bufA, Wh[8], P[17], bufB, R, 256, 512, stream);
        k_maxpool<<<cdiv(B*S*512, TB), TB, 0, stream>>>(bufB, l3f, B*S, K, 512);
    }
    // ================= FP3: (l2x, l3x | l2f, l3f), MLP 768->512->256 -> l2u
    {
        int S1 = 128, S2 = 32, R = B * S1;
        k_interp3<<<cdiv(R, TB), TB, 0, stream>>>(l2x, l3x, l3f, l2f, S1, S2, 512, 256, bufA);
        gemm(bufA, Wh[9],  P[19], bufB, R, 768, 512, stream);
        gemm(bufB, Wh[10], P[21], l2u,  R, 512, 256, stream);
    }
    // ================= FP2: (l1x, l2x | l1f, l2u), MLP 384->256->128 -> l1u
    {
        int S1 = 512, S2 = 128, R = B * S1;
        k_interp3<<<cdiv(R, TB), TB, 0, stream>>>(l1x, l2x, l2u, l1f, S1, S2, 256, 128, bufA);
        gemm(bufA, Wh[11], P[23], bufB, R, 384, 256, stream);
        gemm(bufB, Wh[12], P[25], l1u,  R, 256, 128, stream);
    }
    // ================= FP1: (xyz, l1x | None, l1u), MLP 128->128->256
    {
        int S1 = NPTS, S2 = 512, R = B * S1;
        k_interp3<<<cdiv(R, TB), TB, 0, stream>>>(xyz, l1x, l1u, nullptr, S1, S2, 128, 0, bufA);
        gemm(bufA, Wh[13], P[27], bufB, R, 128, 128, stream);
        gemm(bufB, Wh[14], P[29], bufA, R, 128, 256, stream);
        size_t total = (size_t)B * NPTS * 256;
        k_transpose<<<(int)((total + TB - 1) / TB), TB, 0, stream>>>(bufA, (float*)d_out, NPTS, 256);
    }
}
